// MultiHeadAttention_3246995275965
// MI455X (gfx1250) — compile-verified, matrix operand order checked by reading, not run on hardware
//
#include <hip/hip_runtime.h>
#include <hip/hip_bf16.h>

// ---------------- types ----------------
typedef __attribute__((ext_vector_type(16))) __bf16         v16bf;
typedef __attribute__((ext_vector_type(16))) unsigned short v16us;
typedef __attribute__((ext_vector_type(8)))  float          v8f;

// problem constants
#define BATCH 4
#define SEQ   1024
#define HID   1024
#define NH    16
#define HD    64

// workspace offsets (in ushort/bf16 elements)
#define OFF_X   0u
#define OFF_WQ  4194304u
#define OFF_WK  5242880u
#define OFF_WV  6291456u
#define OFF_Q   7340032u
#define OFF_K   11534336u
#define OFF_VT  15728640u

__device__ __forceinline__ unsigned short f2bf(float f) {
    unsigned u = __builtin_bit_cast(unsigned, f);
    u += 0x7FFFu + ((u >> 16) & 1u);   // round-to-nearest-even
    return (unsigned short)(u >> 16);
}

struct Frag32B { uint4 lo, hi; };

// Loads a 16-half fragment (A-layout chunk): elements p[0..7] -> j=0..7 (K=koff..koff+7),
// p[16..23] -> j=8..15 (K=koff+16..koff+23). Caller bakes koff = 8*(lane>>4) into p.
__device__ __forceinline__ v16bf load_frag(const unsigned short* p) {
    Frag32B f;
    f.lo = *(const uint4*)(p);
    f.hi = *(const uint4*)(p + 16);
    return __builtin_bit_cast(v16bf, f);
}

__device__ __forceinline__ v8f wmma_bf16(v16bf a, v16bf b, v8f c) {
    return __builtin_amdgcn_wmma_f32_16x16x32_bf16(false, a, false, b, (short)0, c, false, false);
}

// ---------------- kernel 0: f32 -> bf16 convert (x, Wq, Wk, Wv) ----------------
__global__ __launch_bounds__(256) void mha_cvt_kernel(const float* __restrict__ x,
                                                      const float* __restrict__ wq,
                                                      const float* __restrict__ wk,
                                                      const float* __restrict__ wv,
                                                      unsigned short* __restrict__ ws) {
    unsigned i = (blockIdx.x * 256u + threadIdx.x) * 4u;   // 7340032 elements total
    const float* src;
    unsigned base;
    if (i < OFF_WQ)      { src = x;  base = OFF_X;  }
    else if (i < OFF_WK) { src = wq; base = OFF_WQ; }
    else if (i < OFF_WV) { src = wk; base = OFF_WK; }
    else                 { src = wv; base = OFF_WV; }
    float4 v = *(const float4*)(src + (i - base));
    ushort4 r;
    r.x = f2bf(v.x); r.y = f2bf(v.y); r.z = f2bf(v.z); r.w = f2bf(v.w);
    *(ushort4*)(ws + i) = r;
}

// ---------------- kernel 1: fused QKV projection (bf16 WMMA, f32 acc) ----------------
// One wave per 16x16 output tile; q/k stored [b,h,s,d] (q pre-scaled by d^-0.5),
// v stored transposed [b,h,d,s] so the attention PV GEMM needs no transpose.
__global__ __launch_bounds__(256) void mha_qkv_kernel(const unsigned short* __restrict__ ws,
                                                      unsigned short* __restrict__ qo,
                                                      unsigned short* __restrict__ ko,
                                                      unsigned short* __restrict__ vto) {
    const int wid   = threadIdx.x >> 5;
    const int lane  = threadIdx.x & 31;
    const int tile  = blockIdx.x * 8 + wid;       // 16384 tiles
    const int ntile = tile & 63;
    const int mtile = tile >> 6;
    const int m0 = mtile * 16, n0 = ntile * 16;
    const int row16 = lane & 15;
    const int lh    = lane >> 4;
    const int koff  = lh * 8;

    const unsigned short* xrow = ws + OFF_X  + (m0 + row16) * HID + koff;
    const unsigned short* qrow = ws + OFF_WQ + (n0 + row16) * HID + koff;
    const unsigned short* krow = ws + OFF_WK + (n0 + row16) * HID + koff;
    const unsigned short* vrow = ws + OFF_WV + (n0 + row16) * HID + koff;

    v8f aq = {}, ak = {}, av = {};
    for (int kk = 0; kk < HID; kk += 32) {
        v16bf a = load_frag(xrow + kk);
        aq = wmma_bf16(a, load_frag(qrow + kk), aq);
        ak = wmma_bf16(a, load_frag(krow + kk), ak);
        av = wmma_bf16(a, load_frag(vrow + kk), av);
    }

    const int b     = m0 >> 10;
    const int sbase = m0 & (SEQ - 1);
    const int h     = n0 >> 6;
    const int d     = (n0 & 63) + row16;          // column n = n0 + (lane&15)
    const int bh    = b * NH + h;
#pragma unroll
    for (int r = 0; r < 8; ++r) {
        int s  = sbase + r + 8 * lh;              // C row m = r + 8*(lane>>4)
        int qi = (bh * SEQ + s) * HD + d;
        qo[qi] = f2bf(aq[r] * 0.125f);            // scaling = 64^-0.5
        ko[qi] = f2bf(ak[r]);
        vto[(bh * HD + d) * SEQ + s] = f2bf(av[r]);
    }
}

// ---------------- kernel 2: flash attention, transposed accumulation ----------------
// Per wave: one (b, h, 16-query tile). Computes S^T = K·Q^T so softmax over keys is an
// in-lane reduction + one shfl_xor(16), and exp(S^T) in C-layout IS the B-fragment for
// O^T = V^T · P^T (no cross-lane shuffles for P).
__global__ __launch_bounds__(256) void mha_attn_kernel(const unsigned short* __restrict__ q,
                                                       const unsigned short* __restrict__ k,
                                                       const unsigned short* __restrict__ vt,
                                                       const float* __restrict__ bias,
                                                       const unsigned char* __restrict__ amask,
                                                       const unsigned char* __restrict__ pmask,
                                                       float* __restrict__ out) {
    const int wid   = threadIdx.x >> 5;
    const int lane  = threadIdx.x & 31;
    const int row16 = lane & 15;
    const int lh    = lane >> 4;
    const int koff  = lh * 8;
    const int h  = blockIdx.y;
    const int b  = blockIdx.z;
    const int s0 = (blockIdx.x * 8 + wid) * 16;
    const int bh = b * NH + h;

    const unsigned short* qptr = q  + (size_t)bh * SEQ * HD;
    const unsigned short* kptr = k  + (size_t)bh * SEQ * HD;
    const unsigned short* vptr = vt + (size_t)bh * HD * SEQ;
    const float*         brow  = bias + ((size_t)bh * SEQ + (s0 + row16)) * SEQ;
    const unsigned char* arow  = amask + (size_t)(s0 + row16) * SEQ;
    const unsigned char* prow  = pmask + (size_t)b * SEQ;

    // Q fragments (reused for all 64 key tiles): B-frag of S^T GEMM = rows of Q
    v16bf qf0 = load_frag(qptr + (s0 + row16) * HD + koff);
    v16bf qf1 = load_frag(qptr + (s0 + row16) * HD + 32 + koff);

    v8f o[4];
#pragma unroll
    for (int f = 0; f < 4; ++f) o[f] = (v8f){};
    float mrow = -__builtin_inff();
    float lrow = 0.0f;

    for (int t0 = 0; t0 < SEQ; t0 += 32) {
        // prefetch the dominant HBM stream (bias) a few iterations ahead
        __builtin_prefetch(brow + t0 + 8 * lh + 256, 0, 0);

        // ---- S^T tiles: rows t (keys), cols s (queries) ----
        v16bf ka0 = load_frag(kptr + (t0 + row16) * HD + koff);
        v16bf ka1 = load_frag(kptr + (t0 + row16) * HD + 32 + koff);
        v16bf kc0 = load_frag(kptr + (t0 + 16 + row16) * HD + koff);
        v16bf kc1 = load_frag(kptr + (t0 + 16 + row16) * HD + 32 + koff);
        v8f zz = {};
        v8f sa = wmma_bf16(ka0, qf0, zz); sa = wmma_bf16(ka1, qf1, sa);  // t0..t0+15
        v8f sb = wmma_bf16(kc0, qf0, zz); sb = wmma_bf16(kc1, qf1, sb);  // t0+16..t0+31

        // ---- bias + masks: lane owns query col s = s0+row16; t is contiguous ----
        const float* bp0 = brow + t0 + 8 * lh;
        const float* bp1 = brow + t0 + 16 + 8 * lh;
        float4 a0 = *(const float4*)(bp0), a1 = *(const float4*)(bp0 + 4);
        float4 c0 = *(const float4*)(bp1), c1 = *(const float4*)(bp1 + 4);
        float bav[8] = {a0.x, a0.y, a0.z, a0.w, a1.x, a1.y, a1.z, a1.w};
        float bbv[8] = {c0.x, c0.y, c0.z, c0.w, c1.x, c1.y, c1.z, c1.w};
        unsigned long long ma = (*(const unsigned long long*)(arow + t0 + 8 * lh)) |
                                (*(const unsigned long long*)(prow + t0 + 8 * lh));
        unsigned long long mb = (*(const unsigned long long*)(arow + t0 + 16 + 8 * lh)) |
                                (*(const unsigned long long*)(prow + t0 + 16 + 8 * lh));
#pragma unroll
        for (int r = 0; r < 8; ++r) {
            float va = sa[r] + bav[r];
            float vb = sb[r] + bbv[r];
            if ((ma >> (8 * r)) & 0xffull) va = -__builtin_inff();
            if ((mb >> (8 * r)) & 0xffull) vb = -__builtin_inff();
            sa[r] = va; sb[r] = vb;
        }

        // ---- online softmax over keys (column of S^T): in-lane + one shfl_xor(16) ----
        float tmax = sa[0];
#pragma unroll
        for (int r = 0; r < 8; ++r) { tmax = fmaxf(tmax, sa[r]); tmax = fmaxf(tmax, sb[r]); }
        tmax = fmaxf(tmax, __shfl_xor(tmax, 16, 32));
        float mnew = fmaxf(mrow, tmax);
        float corr = __expf(mrow - mnew);

        v16us pu;
        float rsum = 0.0f;
#pragma unroll
        for (int r = 0; r < 8; ++r) {
            float ea = __expf(sa[r] - mnew);
            float eb = __expf(sb[r] - mnew);
            rsum += ea + eb;
            pu[r]     = f2bf(ea);   // K = t0 + 8*lh + r        (exact B-frag slot)
            pu[r + 8] = f2bf(eb);   // K = t0 + 16 + 8*lh + r
        }
        lrow = lrow * corr + rsum + __shfl_xor(rsum, 16, 32);
        mrow = mnew;

        // ---- O^T += V^T · P^T : rescale then 4 WMMAs (d tiles of 16) ----
        v16bf p = __builtin_bit_cast(v16bf, pu);
#pragma unroll
        for (int f = 0; f < 4; ++f) {
#pragma unroll
            for (int r = 0; r < 8; ++r) o[f][r] *= corr;
            v16bf vf = load_frag(vptr + (f * 16 + row16) * SEQ + t0 + koff);
            o[f] = wmma_bf16(vf, p, o[f]);
        }
    }

    // ---- epilogue: O^T frag f holds rows d = 16f + r + 8*lh, col s = s0+row16 ----
    float inv = 1.0f / lrow;
    float* orow = out + ((size_t)b * SEQ + s0 + row16) * HID + h * HD;
#pragma unroll
    for (int f = 0; f < 4; ++f) {
        float4 lo4 = make_float4(o[f][0] * inv, o[f][1] * inv, o[f][2] * inv, o[f][3] * inv);
        float4 hi4 = make_float4(o[f][4] * inv, o[f][5] * inv, o[f][6] * inv, o[f][7] * inv);
        *(float4*)(orow + 16 * f + 8 * lh)     = lo4;
        *(float4*)(orow + 16 * f + 8 * lh + 4) = hi4;
    }
}

// ---------------- host launcher ----------------
extern "C" void kernel_launch(void* const* d_in, const int* in_sizes, int n_in,
                              void* d_out, int out_size, void* d_ws, size_t ws_size,
                              hipStream_t stream) {
    const float*         x     = (const float*)d_in[0];
    const float*         bias  = (const float*)d_in[1];
    const unsigned char* amask = (const unsigned char*)d_in[2];
    const unsigned char* pmask = (const unsigned char*)d_in[3];
    const float*         Wq    = (const float*)d_in[4];
    const float*         Wk    = (const float*)d_in[5];
    const float*         Wv    = (const float*)d_in[6];
    unsigned short* ws  = (unsigned short*)d_ws;
    float*          out = (float*)d_out;

    // 7,340,032 f32 elements converted, 4 per thread
    mha_cvt_kernel<<<7168, 256, 0, stream>>>(x, Wq, Wk, Wv, ws);
    // 16384 16x16 output tiles, 8 waves per block
    mha_qkv_kernel<<<2048, 256, 0, stream>>>(ws, ws + OFF_Q, ws + OFF_K, ws + OFF_VT);
    // (8 qtile-groups x 8 waves) x 16 heads x 4 batch
    mha_attn_kernel<<<dim3(8, 16, 4), 256, 0, stream>>>(ws + OFF_Q, ws + OFF_K, ws + OFF_VT,
                                                        bias, amask, pmask, out);
}
